// TextToImageLatentModel_20469814133553
// MI455X (gfx1250) — matrix operand order repaired and checked by reading, not compiled, not run on hardware
//
#include <hip/hip_runtime.h>
#include <hip/hip_bf16.h>
#include <math.h>

// ---------------------------------------------------------------------------
// Transformer block for MI455X (gfx1250, wave32, WMMA + TDM).
// GEMMs: v_wmma_f32_16x16x32_bf16, A panel staged via tensor_load_to_lds.
// ---------------------------------------------------------------------------

constexpr int kE  = 2048;
constexpr int kH  = 16;
constexpr int kD  = 128;      // kE / kH
constexpr int kFF = 8192;
constexpr int kB  = 2;
constexpr int kS  = 2048;
constexpr int kM  = kB * kS;  // 4096 rows
constexpr float kEps = 1e-6f;

typedef __attribute__((ext_vector_type(16))) __bf16    v16bf;
typedef __attribute__((ext_vector_type(8)))  __bf16    v8bf;
typedef __attribute__((ext_vector_type(8)))  float     v8f;
typedef __attribute__((ext_vector_type(4)))  unsigned  v4u;
typedef __attribute__((ext_vector_type(8)))  int       v8i;
typedef __attribute__((ext_vector_type(4)))  int       v4i;

static __device__ __forceinline__ __bf16 f2bf(float f) {
  unsigned u = __builtin_bit_cast(unsigned, f);
  unsigned r = u + 0x7FFFu + ((u >> 16) & 1u);   // round-to-nearest-even
  unsigned short h = (unsigned short)(r >> 16);
  return __builtin_bit_cast(__bf16, h);
}
static __device__ __forceinline__ float bf2f(__bf16 b) {
  unsigned short h = __builtin_bit_cast(unsigned short, b);
  unsigned u = ((unsigned)h) << 16;
  return __builtin_bit_cast(float, u);
}
static __device__ __forceinline__ v8f zero8() {
  v8f z = {0.f, 0.f, 0.f, 0.f, 0.f, 0.f, 0.f, 0.f};
  return z;
}

// A-matrix fragment (16x32 bf16): per lane two contiguous 8-elem chunks.
union AFrag { v16bf v; v8bf h[2]; };

// ---------------------------------------------------------------------------
// TDM: async-load a [rows x 512] bf16 tile (row pitch Kdim elems) into LDS.
// 2D descriptor per CDNA5 ISA ch.8: group0 = {flags, lds_addr, global_addr,
// type=2}; group1 = {data_size=2B, tensor_dim0=Kdim, tensor_dim1=rows,
// tile_dim0=512, tile_dim1=rows, stride0=Kdim}; groups 2/3 zero (<=2D).
// Tracked with TENSORcnt; caller pairs with s_wait_tensorcnt + barrier.
// This toolchain's builtin takes 6 args (extra int32x8 group before cpol).
// ---------------------------------------------------------------------------
static __device__ __forceinline__ void tdm_load_a(const __bf16* gsrc, int Kdim,
                                                  unsigned lds_byte, int rows) {
  unsigned long long ga = (unsigned long long)(size_t)gsrc;
  v4u g0;
  g0[0] = 1u;                                             // count=1, user mode
  g0[1] = lds_byte;                                       // LDS byte address
  g0[2] = (unsigned)ga;                                   // global_addr[31:0]
  g0[3] = (unsigned)((ga >> 32) & 0x01FFFFFFu) | (2u << 30);  // [56:32] | type=2
  v8i g1;
  g1[0] = (int)(1u << 16);                                // data_size = 2 bytes
  g1[1] = (int)((unsigned)Kdim << 16);                    // tensor_dim0[15:0]
  g1[2] = (int)(((unsigned)Kdim >> 16) | ((unsigned)rows << 16)); // dim0 hi | dim1 lo
  g1[3] = (int)(512u << 16);                              // tile_dim0 = 512
  g1[4] = rows;                                           // tile_dim1
  g1[5] = Kdim;                                           // tensor_dim0_stride lo32
  g1[6] = 0;
  g1[7] = 0;
  v4i z4 = {0, 0, 0, 0};
  v8i z8 = {0, 0, 0, 0, 0, 0, 0, 0};
  __builtin_amdgcn_tensor_load_to_lds(g0, g1, z4, z4, z8, 0);
}

// ---------------------------------------------------------------------------
// fp32 -> bf16 conversion (weights)
// ---------------------------------------------------------------------------
__global__ __launch_bounds__(256) void conv_f32_bf16(const float* __restrict__ src,
                                                     __bf16* __restrict__ dst, int n) {
  int i = (blockIdx.x * 256 + threadIdx.x) * 4;
  if (i < n) {
    float4 f = *(const float4*)(src + i);
    dst[i + 0] = f2bf(f.x);
    dst[i + 1] = f2bf(f.y);
    dst[i + 2] = f2bf(f.z);
    dst[i + 3] = f2bf(f.w);
  }
}

// ---------------------------------------------------------------------------
// RMSNorm: one block per row, writes bf16 normalized row
// ---------------------------------------------------------------------------
__global__ __launch_bounds__(256) void rmsnorm_bf16(const float* __restrict__ x,
                                                    const float* __restrict__ w,
                                                    __bf16* __restrict__ out) {
  __shared__ float red[8];
  const int row = blockIdx.x;
  const float* xr = x + (size_t)row * kE;
  float ss = 0.f;
  for (int i = threadIdx.x; i < kE; i += 256) { float v = xr[i]; ss += v * v; }
  #pragma unroll
  for (int m = 16; m >= 1; m >>= 1) ss += __shfl_xor(ss, m, 32);
  if ((threadIdx.x & 31) == 0) red[threadIdx.x >> 5] = ss;
  __syncthreads();
  float tot = 0.f;
  #pragma unroll
  for (int i = 0; i < 8; ++i) tot += red[i];
  const float sc = rsqrtf(tot / (float)kE + kEps);
  __bf16* orow = out + (size_t)row * kE;
  for (int i = threadIdx.x; i < kE; i += 256) orow[i] = f2bf(xr[i] * w[i] * sc);
}

// ---------------------------------------------------------------------------
// Generic WMMA GEMM:  C[M,N] = A[M,K] * W[N,K]^T  (+ fused epilogue)
// Block = 256 thr = 8 waves (2 M-waves x 4 N-waves); wave tile = 16x64;
// block tile = 32x256.  The 32-row A panel is staged through LDS by the
// Tensor Data Mover in double-buffered 512-K chunks (TENSORcnt), so the 4
// N-waves share one copy of A; B streams as pipelined global_load_b128.
// MODE 0: store fp32
// MODE 1: + bias + fp32 residual -> fp32
// MODE 2: + bias, silu -> bf16
// MODE 3: + bias, multiply by bf16 'extra' -> bf16   (SwiGLU product)
// ---------------------------------------------------------------------------
template <int MODE>
__global__ __launch_bounds__(256) void gemm_bf16_wmma(const __bf16* __restrict__ A,
                                                      const __bf16* __restrict__ W,
                                                      const float* __restrict__ bias,
                                                      const void* __restrict__ extra,
                                                      void* __restrict__ out,
                                                      int Ndim, int Kdim) {
  constexpr int KC = 512;                    // K chunk staged per TDM op
  __shared__ __bf16 sA[2][32 * KC];          // 64 KB double buffer (LDS off 0)

  const int lane = threadIdx.x & 31;
  const int wave = threadIdx.x >> 5;
  const int half = lane >> 4;   // 0: lanes 0-15, 1: lanes 16-31
  const int l15  = lane & 15;
  const int mloc = (wave >> 2) * 16;         // 0 or 16 inside the A panel
  const int m0 = blockIdx.x * 32 + mloc;
  const int n0 = blockIdx.y * 256 + (wave & 3) * 64;

  const __bf16* apanel = A + (size_t)blockIdx.x * 32 * Kdim;

  v8f acc[4];
  #pragma unroll
  for (int i = 0; i < 4; ++i) acc[i] = zero8();

  const int nchunks = Kdim / KC;
  if (wave == 0) tdm_load_a(apanel, Kdim, 0u, 32);            // chunk 0

  for (int ci = 0; ci < nchunks; ++ci) {
    if (wave == 0) __builtin_amdgcn_s_wait_tensorcnt(0);      // chunk ci landed
    __syncthreads();                                          // publish to block
    if (wave == 0 && ci + 1 < nchunks)                        // prefetch ci+1
      tdm_load_a(apanel + (size_t)(ci + 1) * KC, Kdim,
                 (unsigned)(((ci + 1) & 1) * 32 * KC * 2), 32);

    const __bf16* aL = &sA[ci & 1][(size_t)(mloc + l15) * KC];
    for (int k = 0; k < KC; k += 32) {
      AFrag af;
      af.h[0] = *(const v8bf*)(aL + k + half * 16);
      af.h[1] = *(const v8bf*)(aL + k + half * 16 + 8);
      const int kB = ci * KC + k + half * 16;
      #pragma unroll
      for (int nt = 0; nt < 4; ++nt) {
        const __bf16* wrow = W + (size_t)(n0 + nt * 16 + l15) * Kdim + kB;
        v16bf bf = *(const v16bf*)wrow;
        acc[nt] = __builtin_amdgcn_wmma_f32_16x16x32_bf16(
            false, af.v, false, bf, (short)0, acc[nt], false, false);
      }
    }
    __syncthreads();        // all waves done with this buffer before reuse
  }

  // Epilogue: C layout — VGPR r, lane L: row = r + (L<16?0:8), col = L&15
  #pragma unroll
  for (int nt = 0; nt < 4; ++nt) {
    #pragma unroll
    for (int r = 0; r < 8; ++r) {
      const int row = m0 + half * 8 + r;
      const int col = n0 + nt * 16 + l15;
      const size_t idx = (size_t)row * Ndim + col;
      float v = acc[nt][r];
      if constexpr (MODE == 0) {
        ((float*)out)[idx] = v;
      } else if constexpr (MODE == 1) {
        ((float*)out)[idx] = v + bias[col] + ((const float*)extra)[idx];
      } else if constexpr (MODE == 2) {
        v += bias[col];
        ((__bf16*)out)[idx] = f2bf(v / (1.0f + __expf(-v)));   // silu
      } else {
        v += bias[col];
        ((__bf16*)out)[idx] = f2bf(v * bf2f(((const __bf16*)extra)[idx]));
      }
    }
  }
}

// ---------------------------------------------------------------------------
// RoPE + head split.  qkv fp32 [M, 3E] -> q,k bf16 [bh, s, d], v bf16 [bh, d, s]
// q is pre-multiplied by 1/sqrt(D) so the attention hot loop has no scaling.
// ---------------------------------------------------------------------------
__global__ __launch_bounds__(256) void rope_split(const float* __restrict__ qkv,
                                                  __bf16* __restrict__ qb,
                                                  __bf16* __restrict__ kb,
                                                  __bf16* __restrict__ vt) {
  const int t = blockIdx.x * 256 + threadIdx.x;     // [0, M*H*D/2)
  const int HD2 = kH * (kD / 2);
  const int m = t / HD2;
  const int rem = t - m * HD2;
  const int h = rem / (kD / 2);
  const int i = rem - h * (kD / 2);
  const int b = m / kS, s = m - b * kS;
  const float scale = 0.08838834764831845f;         // 1/sqrt(128), folded into q

  const float fr = (float)s * __powf(10000.0f, -(float)(2 * i) / (float)kD);
  float sn, c;
  __sincosf(fr, &sn, &c);

  const float* base = qkv + (size_t)m * (3 * kE) + h * kD + 2 * i;
  const float qe = base[0],        qo = base[1];
  const float ke = base[kE],       ko = base[kE + 1];
  const float v0 = base[2 * kE],   v1 = base[2 * kE + 1];

  const size_t bh = (size_t)(b * kH + h);
  const size_t qidx = (bh * kS + s) * kD;
  qb[qidx + i]          = f2bf((qe * c  - qo * sn) * scale);
  qb[qidx + kD / 2 + i] = f2bf((qe * sn + qo * c)  * scale);
  kb[qidx + i]          = f2bf(ke * c  - ko * sn);
  kb[qidx + kD / 2 + i] = f2bf(ke * sn + ko * c);
  const size_t vbase = bh * kD * kS;
  vt[vbase + (size_t)(2 * i)     * kS + s] = f2bf(v0);
  vt[vbase + (size_t)(2 * i + 1) * kS + s] = f2bf(v1);
}

// ---------------------------------------------------------------------------
// Flash attention: one wave = 16 query rows; online softmax over 32-key chunks.
// scores: 8 WMMAs (K=128 contraction, two 16-key tiles); PV: 8 WMMAs (D=128).
// P re-shaped C-layout -> A-layout through 1KB of LDS per wave.
// ---------------------------------------------------------------------------
__global__ __launch_bounds__(128) void flash_attn(const __bf16* __restrict__ qb,
                                                  const __bf16* __restrict__ kb,
                                                  const __bf16* __restrict__ vt,
                                                  float* __restrict__ out) {
  __shared__ __bf16 plds[4][16 * 32];
  const int lane = threadIdx.x & 31;
  const int wave = threadIdx.x >> 5;
  const int half = lane >> 4;
  const int l15  = lane & 15;
  const int bh = blockIdx.y;
  const int b = bh / kH, h = bh - b * kH;
  const int q0 = blockIdx.x * 64 + wave * 16;

  // Q fragments for the 4 K-chunks of 32 (contraction dim = D = 128)
  const __bf16* qrow = qb + ((size_t)bh * kS + q0 + l15) * kD;
  v16bf qf[4];
  #pragma unroll
  for (int c = 0; c < 4; ++c) {
    AFrag u;
    u.h[0] = *(const v8bf*)(qrow + c * 32 + half * 16);
    u.h[1] = *(const v8bf*)(qrow + c * 32 + half * 16 + 8);
    qf[c] = u.v;
  }

  v8f oacc[8];
  #pragma unroll
  for (int i = 0; i < 8; ++i) oacc[i] = zero8();
  v8f mrun, lrun;
  #pragma unroll
  for (int r = 0; r < 8; ++r) { mrun[r] = -1e30f; lrun[r] = 0.f; }

  const __bf16* kbase = kb + (size_t)bh * kS * kD;
  const __bf16* vbase = vt + (size_t)bh * kD * kS;
  __bf16* myp = &plds[wave][0];

  for (int kt = 0; kt < kS; kt += 32) {
    // ---- scores for keys [kt, kt+32) : two 16-key tiles (scale pre-folded)
    v8f s0 = zero8(), s1 = zero8();
    const __bf16* k0 = kbase + (size_t)(kt + l15) * kD + half * 16;
    const __bf16* k1 = k0 + (size_t)16 * kD;
    #pragma unroll
    for (int c = 0; c < 4; ++c) {
      v16bf b0 = *(const v16bf*)(k0 + c * 32);
      s0 = __builtin_amdgcn_wmma_f32_16x16x32_bf16(false, qf[c], false, b0, (short)0, s0, false, false);
      v16bf b1 = *(const v16bf*)(k1 + c * 32);
      s1 = __builtin_amdgcn_wmma_f32_16x16x32_bf16(false, qf[c], false, b1, (short)0, s1, false, false);
    }

    // ---- online softmax update (row stats reduced across each 16-lane half)
    v8f t;
    #pragma unroll
    for (int r = 0; r < 8; ++r) t[r] = fmaxf(s0[r], s1[r]);
    #pragma unroll
    for (int msk = 1; msk <= 8; msk <<= 1)
      #pragma unroll
      for (int r = 0; r < 8; ++r)
        t[r] = fmaxf(t[r], __shfl_xor(t[r], msk, 32));

    v8f alpha, ps;
    #pragma unroll
    for (int r = 0; r < 8; ++r) {
      const float mn = fmaxf(mrun[r], t[r]);
      alpha[r] = __expf(mrun[r] - mn);
      mrun[r] = mn;
      s0[r] = __expf(s0[r] - mn);
      s1[r] = __expf(s1[r] - mn);
      ps[r] = s0[r] + s1[r];
    }
    #pragma unroll
    for (int msk = 1; msk <= 8; msk <<= 1)
      #pragma unroll
      for (int r = 0; r < 8; ++r)
        ps[r] += __shfl_xor(ps[r], msk, 32);
    #pragma unroll
    for (int r = 0; r < 8; ++r) lrun[r] = lrun[r] * alpha[r] + ps[r];
    #pragma unroll
    for (int nt = 0; nt < 8; ++nt)
      #pragma unroll
      for (int r = 0; r < 8; ++r)
        oacc[nt][r] *= alpha[r];

    // ---- stage P (16x32 bf16) through LDS: C-layout -> A-layout
    #pragma unroll
    for (int r = 0; r < 8; ++r) {
      const int Mr = half * 8 + r;
      myp[Mr * 32 + l15]      = f2bf(s0[r]);
      myp[Mr * 32 + 16 + l15] = f2bf(s1[r]);
    }
    __syncthreads();
    AFrag pu;
    pu.h[0] = *(const v8bf*)(myp + l15 * 32 + half * 16);
    pu.h[1] = *(const v8bf*)(myp + l15 * 32 + half * 16 + 8);

    // ---- P (16x32) x V (32x128): v stored [d, s] so per-lane loads contiguous
    #pragma unroll
    for (int nt = 0; nt < 8; ++nt) {
      v16bf bv = *(const v16bf*)(vbase + (size_t)(nt * 16 + l15) * kS + kt + half * 16);
      oacc[nt] = __builtin_amdgcn_wmma_f32_16x16x32_bf16(false, pu.v, false, bv, (short)0, oacc[nt], false, false);
    }
  }

  // ---- finalize: out[b, s, h*D + d] fp32
  #pragma unroll
  for (int r = 0; r < 8; ++r) lrun[r] = 1.0f / lrun[r];
  #pragma unroll
  for (int nt = 0; nt < 8; ++nt)
    #pragma unroll
    for (int r = 0; r < 8; ++r) {
      const int srow = q0 + half * 8 + r;
      const int col = h * kD + nt * 16 + l15;
      out[((size_t)b * kS + srow) * kE + col] = oacc[nt][r] * lrun[r];
    }
}

// ---------------------------------------------------------------------------
// og = bf16( attn_out * sigmoid(gate_raw) )
// ---------------------------------------------------------------------------
__global__ __launch_bounds__(256) void gate_mul(const float* __restrict__ attn,
                                                const float* __restrict__ gate,
                                                __bf16* __restrict__ og, int n) {
  int i = blockIdx.x * 256 + threadIdx.x;
  if (i < n) {
    const float g = 1.0f / (1.0f + __expf(-gate[i]));
    og[i] = f2bf(attn[i] * g);
  }
}

// ---------------------------------------------------------------------------
// Host orchestration
// ---------------------------------------------------------------------------
extern "C" void kernel_launch(void* const* d_in, const int* in_sizes, int n_in,
                              void* d_out, int out_size, void* d_ws, size_t ws_size,
                              hipStream_t stream) {
  (void)in_sizes; (void)n_in; (void)out_size; (void)ws_size;
  const float* x     = (const float*)d_in[0];
  const float* rms_w = (const float*)d_in[1];
  const float* wqkv  = (const float*)d_in[2];
  const float* wgate = (const float*)d_in[3];
  const float* wout  = (const float*)d_in[4];
  const float* bout  = (const float*)d_in[5];
  const float* w1    = (const float*)d_in[6];
  const float* b1    = (const float*)d_in[7];
  const float* w3    = (const float*)d_in[8];
  const float* b3    = (const float*)d_in[9];
  const float* w2    = (const float*)d_in[10];
  const float* b2    = (const float*)d_in[11];

  char* ws = (char*)d_ws;
  size_t off = 0;
  auto alloc = [&](size_t bytes) {
    char* p = ws + off;
    off += (bytes + 255) & ~(size_t)255;
    return p;
  };

  // bf16 weights: 143 MB total -> resident in 192 MB L2 across all GEMMs
  __bf16* wqkv_b  = (__bf16*)alloc((size_t)3 * kE * kE * 2);
  __bf16* wgate_b = (__bf16*)alloc((size_t)kE * kE * 2);
  __bf16* wout_b  = (__bf16*)alloc((size_t)kE * kE * 2);
  __bf16* w1_b    = (__bf16*)alloc((size_t)kFF * kE * 2);
  __bf16* w3_b    = (__bf16*)alloc((size_t)kFF * kE * 2);
  __bf16* w2_b    = (__bf16*)alloc((size_t)kE * kFF * 2);
  __bf16* xn_b    = (__bf16*)alloc((size_t)kM * kE * 2);
  __bf16* yn_b    = (__bf16*)alloc((size_t)kM * kE * 2);
  __bf16* q_b     = (__bf16*)alloc((size_t)kM * kE * 2);
  __bf16* k_b     = (__bf16*)alloc((size_t)kM * kE * 2);
  __bf16* v_t     = (__bf16*)alloc((size_t)kM * kE * 2);
  float*  x1_f    = (float*) alloc((size_t)kM * kE * 4);
  float*  attn_f  = (float*) alloc((size_t)kM * kE * 4);
  __bf16* og_b    = (__bf16*)alloc((size_t)kM * kE * 2);
  // Big scratch A: qkv fp32 (100.7 MB), later silu(h1) bf16 (67 MB)
  char* bigA = alloc((size_t)kM * 3 * kE * 4);
  // Big scratch B: gate fp32 (33.6 MB), later SwiGLU product bf16 (67 MB)
  char* bigB = alloc((size_t)kM * kFF * 2);
  float*  qkv_f  = (float*)bigA;
  __bf16* s1_b   = (__bf16*)bigA;
  float*  gate_f = (float*)bigB;
  __bf16* hp_b   = (__bf16*)bigB;

  // 1) weight conversion (fp32 -> bf16)
  auto conv = [&](const float* s, __bf16* d, int n) {
    conv_f32_bf16<<<n / 1024, 256, 0, stream>>>(s, d, n);
  };
  conv(wqkv,  wqkv_b,  3 * kE * kE);
  conv(wgate, wgate_b, kE * kE);
  conv(wout,  wout_b,  kE * kE);
  conv(w1,    w1_b,    kFF * kE);
  conv(w3,    w3_b,    kFF * kE);
  conv(w2,    w2_b,    kE * kFF);

  // 2) xn = rmsnorm(x)
  rmsnorm_bf16<<<kM, 256, 0, stream>>>(x, rms_w, xn_b);

  // 3) qkv = xn @ wqkv^T  (fp32 out, rope needs full precision pairs)
  dim3 gqkv(kM / 32, (3 * kE) / 256);
  gemm_bf16_wmma<0><<<gqkv, 256, 0, stream>>>(xn_b, wqkv_b, nullptr, nullptr, qkv_f, 3 * kE, kE);

  // 4) rope + head split (+ v transpose, q pre-scaled)
  rope_split<<<(kM * kH * (kD / 2)) / 256, 256, 0, stream>>>(qkv_f, q_b, k_b, v_t);

  // 5) gate_raw = xn @ w_gate^T
  dim3 gE(kM / 32, kE / 256);
  gemm_bf16_wmma<0><<<gE, 256, 0, stream>>>(xn_b, wgate_b, nullptr, nullptr, gate_f, kE, kE);

  // 6) attention (flash, online softmax)
  dim3 ga(kS / 64, kB * kH);
  flash_attn<<<ga, 128, 0, stream>>>(q_b, k_b, v_t, attn_f);

  // 7) og = attn_out * sigmoid(gate_raw)
  gate_mul<<<(kM * kE) / 256, 256, 0, stream>>>(attn_f, gate_f, og_b, kM * kE);

  // 8) x1 = x + og @ w_out^T + b_out
  gemm_bf16_wmma<1><<<gE, 256, 0, stream>>>(og_b, wout_b, bout, x, x1_f, kE, kE);

  // 9) yn = rmsnorm(x1)
  rmsnorm_bf16<<<kM, 256, 0, stream>>>(x1_f, rms_w, yn_b);

  // 10) s1 = silu(yn @ w1^T + b1)   (bf16)
  dim3 gFF(kM / 32, kFF / 256);
  gemm_bf16_wmma<2><<<gFF, 256, 0, stream>>>(yn_b, w1_b, b1, nullptr, s1_b, kFF, kE);

  // 11) hp = s1 * (yn @ w3^T + b3)  (bf16)
  gemm_bf16_wmma<3><<<gFF, 256, 0, stream>>>(yn_b, w3_b, b3, s1_b, hp_b, kFF, kE);

  // 12) out = x1 + hp @ w2^T + b2
  gemm_bf16_wmma<1><<<gE, 256, 0, stream>>>(hp_b, w2_b, b2, x1_f, (float*)d_out, kE, kFF);
}
